// RFAutoEncoderDecoderWithAttention_89472758710899
// MI455X (gfx1250) — matrix operand order repaired and checked by reading, not compiled, not run on hardware
//
#include <hip/hip_runtime.h>

#define D_    1024
#define DFF_  4096
#define RF_   16
#define NKV_  1152
#define NROW_ 18432
#define CHUNK 2048   // DFF processed in 2 chunks to cap scratch use
#define ALD   72     // padded LDS row stride (elements) for staged A tile
#define ABUF  (128 * ALD)

#define USE_ASYNC_LDS 1

typedef __attribute__((ext_vector_type(16))) __bf16 bf16x16;
typedef __attribute__((ext_vector_type(8)))  __bf16 bf16x8;
typedef __attribute__((ext_vector_type(8)))  float  f32x8;

static __device__ __forceinline__ f32x8 wmma_bf16(bf16x16 a, bf16x16 b, f32x8 c) {
  return __builtin_amdgcn_wmma_f32_16x16x32_bf16(false, a, false, b, (short)0, c, false, false);
}

// A-fragment for 16-bit A 16x32, contiguous k order (this code's choice, not the ISA table order):
//   lane l (row = l&15): elems 0..15 -> K=(l<16?0:16)+j (contiguous, matches B)
static __device__ __forceinline__ bf16x16 frag_a(const __bf16* base, int ld, int r, int half, int k0) {
  bf16x8 lo = *(const bf16x8*)(base + (size_t)r * ld + k0 + half * 16);
  bf16x8 hi = *(const bf16x8*)(base + (size_t)r * ld + k0 + half * 16 + 8);
  return __builtin_shufflevector(lo, hi, 0,1,2,3,4,5,6,7,8,9,10,11,12,13,14,15);
}

static __device__ __forceinline__ float gelu_exact(float x) {
  return 0.5f * x * (1.0f + erff(x * 0.7071067811865476f));
}

#if USE_ASYNC_LDS
static __device__ __forceinline__ void async_copy16(unsigned lds_off, const void* gptr) {
  // GLOBAL_LOAD_ASYNC_TO_LDS_B128: per-lane 16B global -> LDS, tracked by ASYNCcnt
  asm volatile("global_load_async_to_lds_b128 %0, %1, off" :: "v"(lds_off), "v"(gptr) : "memory");
}
static __device__ __forceinline__ void wait_async() {
  asm volatile("s_wait_asynccnt 0x0" ::: "memory");
}
#endif

// ---------- transpose + f32 -> bf16: in (K x N) row-major -> out (N x K) row-major ----------
extern "C" __global__ void k_transpose_cast(const float* __restrict__ in, __bf16* __restrict__ out,
                                            int K, int N) {
  __shared__ float tile[32][33];
  const int nb = blockIdx.x * 32, kb = blockIdx.y * 32;
  const int tx = threadIdx.x, ty = threadIdx.y;   // (32, 8)
  #pragma unroll
  for (int i = ty; i < 32; i += 8) tile[i][tx] = in[(size_t)(kb + i) * N + nb + tx];
  __syncthreads();
  #pragma unroll
  for (int i = ty; i < 32; i += 8) out[(size_t)(nb + i) * K + kb + tx] = (__bf16)tile[tx][i];
}

// ---------- plain f32 -> bf16 cast ----------
extern "C" __global__ void k_cast_bf16(const float* __restrict__ in, __bf16* __restrict__ out, int n) {
  int i = blockIdx.x * 256 + threadIdx.x;
  if (i < n) out[i] = (__bf16)in[i];
}

// ---------- generic 128x128-tile WMMA GEMM, double-buffered async A staging ----------
// C[M x Ncols] = A(M x K bf16, ld=lda) @ Bt^T (+bias), Bt row n holds K values at (n*ldb + koff).
// mode 0: outb = bf16(gelu(val))   mode 1: outf += val
// mode 2: outb = bf16(val)         mode 3: outf  = val
extern "C" __global__ void __launch_bounds__(256)
k_mm(const __bf16* __restrict__ A, int lda,
     const __bf16* __restrict__ Bt, int ldb, int koff, int K,
     const float* __restrict__ bias,
     __bf16* __restrict__ outb, float* __restrict__ outf, int ldo,
     int mode) {
  __shared__ __bf16 As[2 * ABUF];
  const int tid = threadIdx.x;
  const int w = tid >> 5, l = tid & 31, r = l & 15, half = l >> 4;
  const int rowbase = blockIdx.x * 128;
  const int n0 = blockIdx.y * 128 + w * 16;

  const f32x8 zero8 = {0.f,0.f,0.f,0.f,0.f,0.f,0.f,0.f};
  f32x8 acc[8];
  #pragma unroll
  for (int mt = 0; mt < 8; ++mt) acc[mt] = zero8;

  const int srow = tid >> 1, scol = (tid & 1) * 16;      // staging: 2 threads per row, 64 K-cols
  const __bf16* Arow = A + (size_t)(rowbase + srow) * lda + scol;
  const __bf16* brow = Bt + (size_t)(n0 + r) * ldb + koff + half * 16;
#if USE_ASYNC_LDS
  const unsigned lds0 = (unsigned)(size_t)(As + srow * ALD + scol);  // LDS offset (low 32 bits)
#endif

  const int nk = K / 64;
  // prologue: stage chunk 0 into buffer 0
#if USE_ASYNC_LDS
  async_copy16(lds0,      Arow);
  async_copy16(lds0 + 64, Arow + 32);
#else
  *(bf16x16*)(As + srow * ALD + scol)      = *(const bf16x16*)(Arow);
  *(bf16x16*)(As + srow * ALD + scol + 32) = *(const bf16x16*)(Arow + 32);
#endif

  for (int kc = 0; kc < nk; ++kc) {
#if USE_ASYNC_LDS
    wait_async();            // my copies (for chunk kc) are in LDS
#endif
    __syncthreads();         // everyone's copies landed; everyone done reading other buffer
    const int cur = kc & 1;
    if (kc + 1 < nk) {       // stage chunk kc+1 into the other buffer (overlaps with WMMAs below)
      const int k0n = (kc + 1) * 64;
#if USE_ASYNC_LDS
      const unsigned dst = lds0 + (unsigned)((1 - cur) * ABUF * 2);
      async_copy16(dst,      Arow + k0n);
      async_copy16(dst + 64, Arow + k0n + 32);
#else
      __bf16* sb = As + (1 - cur) * ABUF + srow * ALD + scol;
      *(bf16x16*)(sb)      = *(const bf16x16*)(Arow + k0n);
      *(bf16x16*)(sb + 32) = *(const bf16x16*)(Arow + k0n + 32);
#endif
    }
    __builtin_prefetch(brow + kc * 64 + 128, 0, 1);
    const __bf16* Ab = As + cur * ABUF;
    const int k0 = kc * 64;
    #pragma unroll
    for (int ks = 0; ks < 2; ++ks) {
      bf16x16 b = *(const bf16x16*)(brow + k0 + ks * 32);
      bf16x16 af[8];
      #pragma unroll
      for (int mt = 0; mt < 8; ++mt)
        af[mt] = frag_a(Ab + mt * 16 * ALD, ALD, r, half, ks * 32);
      #pragma unroll
      for (int mt = 0; mt < 8; ++mt)
        acc[mt] = wmma_bf16(af[mt], b, acc[mt]);
    }
  }

  const int col = n0 + r;
  const float bval = bias ? bias[col] : 0.f;
  #pragma unroll
  for (int mt = 0; mt < 8; ++mt) {
    #pragma unroll
    for (int v = 0; v < 8; ++v) {
      const size_t row = (size_t)rowbase + mt * 16 + v + half * 8;  // C layout: half -> M 0..7 / 8..15
      const float val = acc[mt][v] + bval;
      if      (mode == 0) outb[row * ldo + col] = (__bf16)gelu_exact(val);
      else if (mode == 1) outf[row * ldo + col] += val;
      else if (mode == 2) outb[row * ldo + col] = (__bf16)val;
      else                outf[row * ldo + col] = val;
    }
  }
}

// ---------- residual add + LayerNorm -> bf16 XN (and updated x) ----------
extern "C" __global__ void __launch_bounds__(256)
k_ln(const float* __restrict__ rfq, const float* __restrict__ addv,
     const float* __restrict__ lns, const float* __restrict__ lnb,
     float* __restrict__ X, __bf16* __restrict__ XN, int first_layer) {
  const int tid = threadIdx.x;
  const int r2 = tid >> 4, sub = tid & 15;
  const int rb = blockIdx.x;                       // n index (RF == 16)
  float* xrow = X + ((size_t)rb * 16 + r2) * D_;
  __bf16* nrow = XN + ((size_t)rb * 16 + r2) * D_;
  const float* arow = addv + (size_t)rb * D_;
  float vals[64];
  float s = 0.f, ss = 0.f;
  if (first_layer) {
    const float* qrow = rfq + (size_t)r2 * D_;
    #pragma unroll 8
    for (int j = 0; j < 64; ++j) {
      int c = sub + j * 16;
      float xv = qrow[c] + arow[c];
      vals[j] = xv; xrow[c] = xv; s += xv; ss += xv * xv;
    }
  } else {
    #pragma unroll 8
    for (int j = 0; j < 64; ++j) {
      int c = sub + j * 16;
      float xv = xrow[c] + arow[c];
      vals[j] = xv; xrow[c] = xv; s += xv; ss += xv * xv;
    }
  }
  #pragma unroll
  for (int o = 8; o > 0; o >>= 1) { s += __shfl_xor(s, o, 16); ss += __shfl_xor(ss, o, 16); }
  const float mean = s * (1.f / D_);
  const float rstd = rsqrtf(ss * (1.f / D_) - mean * mean + 1e-5f);
  #pragma unroll 8
  for (int j = 0; j < 64; ++j) {
    int c = sub + j * 16;
    nrow[c] = (__bf16)((vals[j] - mean) * rstd * lns[c] + lnb[c]);
  }
}

// ---------- final LayerNorm in place on d_out ----------
extern "C" __global__ void __launch_bounds__(256)
k_final_ln(float* __restrict__ X, const float* __restrict__ s, const float* __restrict__ b) {
  const int tid = threadIdx.x;
  const int r2 = tid >> 4, sub = tid & 15;
  float* xrow = X + ((size_t)blockIdx.x * 16 + r2) * D_;
  float vals[64];
  float sm = 0.f, ssq = 0.f;
  #pragma unroll 8
  for (int j = 0; j < 64; ++j) {
    int c = sub + j * 16;
    float v = xrow[c]; vals[j] = v; sm += v; ssq += v * v;
  }
  #pragma unroll
  for (int o = 8; o > 0; o >>= 1) { sm += __shfl_xor(sm, o, 16); ssq += __shfl_xor(ssq, o, 16); }
  const float mean = sm * (1.f / D_);
  const float rstd = rsqrtf(ssq * (1.f / D_) - mean * mean + 1e-5f);
  #pragma unroll 8
  for (int j = 0; j < 64; ++j) {
    int c = sub + j * 16;
    xrow[c] = (vals[j] - mean) * rstd * s[c] + b[c];
  }
}

extern "C" void kernel_launch(void* const* d_in, const int* in_sizes, int n_in,
                              void* d_out, int out_size, void* d_ws, size_t ws_size,
                              hipStream_t stream) {
  (void)in_sizes; (void)n_in; (void)out_size; (void)ws_size;
  const float* compressed = (const float*)d_in[0];
  const float* rfq  = (const float*)d_in[1];
  // d_in[2..7]: ln1/wq/bq/wk/bk are dead (softmax over kv_len==1 is identically 1)
  const float* wv   = (const float*)d_in[8];
  const float* bv   = (const float*)d_in[9];
  const float* wo   = (const float*)d_in[10];
  const float* bo   = (const float*)d_in[11];
  const float* ln2s = (const float*)d_in[12];
  const float* ln2b = (const float*)d_in[13];
  const float* w1   = (const float*)d_in[14];
  const float* b1   = (const float*)d_in[15];
  const float* w2   = (const float*)d_in[16];
  const float* b2   = (const float*)d_in[17];
  const float* lnfs = (const float*)d_in[18];
  const float* lnfb = (const float*)d_in[19];
  float* X = (float*)d_out;                 // x lives in d_out; final LN in place

  char* wsp = (char*)d_ws; size_t off = 0;
  auto carve = [&](size_t bytes) -> char* {
    char* p = wsp + off; off += (bytes + 255) & ~(size_t)255; return p;
  };
  __bf16* wvt  = (__bf16*)carve((size_t)D_ * D_ * 2);          // per-layer, reused
  __bf16* wot  = (__bf16*)carve((size_t)D_ * D_ * 2);
  __bf16* w1t  = (__bf16*)carve((size_t)D_ * DFF_ * 2);
  __bf16* w2t  = (__bf16*)carve((size_t)D_ * DFF_ * 2);
  __bf16* kvb  = (__bf16*)carve((size_t)NKV_ * D_ * 2);
  __bf16* vb   = (__bf16*)carve((size_t)NKV_ * D_ * 2);
  float*  addv = (float*) carve((size_t)NKV_ * D_ * 4);
  __bf16* XN   = (__bf16*)carve((size_t)NROW_ * D_ * 2);
  __bf16* Gc   = (__bf16*)carve((size_t)NROW_ * CHUNK * 2);

  const dim3 tb(32, 8);
  k_cast_bf16<<<(NKV_ * D_ + 255) / 256, 256, 0, stream>>>(compressed, kvb, NKV_ * D_);

  for (int l2 = 0; l2 < 2; ++l2) {
    // per-layer weight transpose+convert (bf16, N-major)
    k_transpose_cast<<<dim3(D_/32,   D_/32),   tb, 0, stream>>>(wv + (size_t)l2*D_*D_,   wvt, D_,   D_);
    k_transpose_cast<<<dim3(D_/32,   D_/32),   tb, 0, stream>>>(wo + (size_t)l2*D_*D_,   wot, D_,   D_);
    k_transpose_cast<<<dim3(DFF_/32, D_/32),   tb, 0, stream>>>(w1 + (size_t)l2*D_*DFF_, w1t, D_,   DFF_);
    k_transpose_cast<<<dim3(D_/32,   DFF_/32), tb, 0, stream>>>(w2 + (size_t)l2*D_*DFF_, w2t, DFF_, D_);

    // attention collapses to: addv[n] = (kv[n]@wv + bv) @ wo + bo
    k_mm<<<dim3(NKV_/128, D_/128), 256, 0, stream>>>(
        kvb, D_, wvt, D_, 0, D_, bv + (size_t)l2*D_, vb, nullptr, D_, 2);
    k_mm<<<dim3(NKV_/128, D_/128), 256, 0, stream>>>(
        vb, D_, wot, D_, 0, D_, bo + (size_t)l2*D_, nullptr, addv, D_, 3);

    // x += addv (bcast over RF); XN = bf16(LN2(x))
    k_ln<<<dim3(NKV_), 256, 0, stream>>>(
        rfq, addv, ln2s + (size_t)l2*D_, ln2b + (size_t)l2*D_, X, XN, l2 == 0);

    // MLP in two DFF chunks: Gc = gelu(XN@w1 + b1)[:, chunk]; x += Gc @ w2[chunk, :] (+b2 once)
    for (int c = 0; c < DFF_ / CHUNK; ++c) {
      k_mm<<<dim3(NROW_/128, CHUNK/128), 256, 0, stream>>>(
          XN, D_, w1t + (size_t)c*CHUNK*D_, D_, 0, D_,
          b1 + (size_t)l2*DFF_ + c*CHUNK, Gc, nullptr, CHUNK, 0);
      k_mm<<<dim3(NROW_/128, D_/128), 256, 0, stream>>>(
          Gc, CHUNK, w2t, DFF_, c*CHUNK, CHUNK,
          (c == 0) ? (b2 + (size_t)l2*D_) : nullptr, nullptr, X, D_, 1);
    }
  }
  k_final_ln<<<dim3(NROW_/16), 256, 0, stream>>>(X, lnfs, lnfb);
}
